// GateV3_34935263986342
// MI455X (gfx1250) — compile-verified, matrix operand order checked by reading, not run on hardware
//
#include <hip/hip_runtime.h>
#include <hip/hip_bf16.h>
#include <stdint.h>

// ---------------------------------------------------------------------------
// MoE router gate (DeepSeek-V3 style) for MI455X / gfx1250.
//   logits = x @ W^T          : [16384,4096] x [256,4096]^T  (bf16 WMMA, f32 acc)
//   softmax -> group-limited top-4 groups -> top-8 experts -> renorm
// Roofline: 34.4 GFLOP, 268 MB mandatory x traffic -> ~11.5 us floor @ 23.3 TB/s.
// bf16 WMMA (16x16x32) keeps the GEMM at the memory floor. 256 threads/8 waves
// (2 waves/SIMD). The A (HBM) stream is register-prefetched one chunk ahead
// under the WMMA block. The B (L2-resident weight) stream uses the CDNA5
// async global->LDS path (ASYNCcnt) when available, eliminating its 32-VGPR
// staging round-trip and freeing register slack for fragment pipelining.
// ---------------------------------------------------------------------------

#define DIMK    4096
#define NEXP    256
#define TOKENS  16384
#define TOPK    8
#define NGROUPS 8
#define NLIM    4

typedef __attribute__((ext_vector_type(8)))  float  v8f;
typedef __attribute__((ext_vector_type(16))) __bf16 v16bf;
typedef int v4i_b128 __attribute__((vector_size(16)));   // matches builtin param

#if __has_builtin(__builtin_amdgcn_global_load_async_to_lds_b128) && \
    __has_builtin(__builtin_amdgcn_s_wait_asynccnt)
#define USE_ASYNC_B 1
#else
#define USE_ASYNC_B 0
#endif

union Frag32B {
    uint4 u[2];
    v16bf v;
};

__device__ __forceinline__ unsigned short f32_to_bf16_rne(float f) {
    unsigned u = __float_as_uint(f);
    u = (u + 0x7FFFu + ((u >> 16) & 1u)) >> 16;
    return (unsigned short)u;
}

// ---------------------------------------------------------------------------
// Kernel 1: weight fp32 -> bf16 (1M elements, 4 per thread)
// ---------------------------------------------------------------------------
__global__ void cvt_weight_kernel(const float* __restrict__ w,
                                  unsigned short* __restrict__ wbf) {
    int i = (blockIdx.x * 256 + threadIdx.x) * 4;
    float4 f = *(const float4*)(w + i);
    ushort4 r;
    r.x = f32_to_bf16_rne(f.x);
    r.y = f32_to_bf16_rne(f.y);
    r.z = f32_to_bf16_rne(f.z);
    r.w = f32_to_bf16_rne(f.w);
    *(ushort4*)(wbf + i) = r;
}

// ---------------------------------------------------------------------------
// Kernel 2: GEMM. Block = 128 tokens x 256 experts, 256 threads / 8 waves;
// wave w owns the 16-token strip [16w,16w+16) x all 256 experts = 16 WMMA
// tiles (128 f32 acc VGPRs). K streamed in chunks of 64 through LDS (row
// stride padded to 72 bf16 = conflict-free for the fragment pattern).
// ---------------------------------------------------------------------------
#define KC 64
#define LDSTR 72   // 72 bf16 = 144 B row stride (16B-aligned, conflict-free)
#define NCHUNK (DIMK / KC)

__global__ void __launch_bounds__(256) router_gemm_kernel(
        const float* __restrict__ x,
        const unsigned short* __restrict__ wbf,
        float* __restrict__ logits) {
    __shared__ unsigned short Abuf[128][LDSTR];
    __shared__ unsigned short Bbuf[256][LDSTR];

    const int tid  = threadIdx.x;
    const int lane = tid & 31;
    const int wid  = tid >> 5;
    const size_t tokBase = (size_t)blockIdx.x * 128;

    v8f acc[16];
#pragma unroll
    for (int t = 0; t < 16; ++t) acc[t] = (v8f)0.0f;

    const int m16 = lane & 15;         // row within strip / expert within tile
    const int kh  = (lane >> 4) * 8;   // half-wave K offset per ISA 16-bit A layout

    // Per-thread staging indices (256 threads):
    //   A: 128 rows x 16 float4 -> 2048 float4, 8 per thread
    //   B: 256 rows x  8 uint4  -> 2048 uint4,  8 per thread
    int aRow[8], aCol[8], bRow[8], bCol[8];
#pragma unroll
    for (int i = 0; i < 8; ++i) {
        int ia = tid + i * 256;
        aRow[i] = ia >> 4;
        aCol[i] = (ia & 15) * 4;
        bRow[i] = ia >> 3;
        bCol[i] = (ia & 7) * 8;
    }

    // --- prologue: register-load A chunk 0 (HBM stream) ---
    float4 aReg[8];
#pragma unroll
    for (int i = 0; i < 8; ++i)
        aReg[i] = *(const float4*)(x + (tokBase + aRow[i]) * DIMK + aCol[i]);

#pragma unroll 1
    for (int c = 0; c < NCHUNK; ++c) {
        const int k0 = c * KC;

        // --- B tile: async L2->LDS (no VGPR round-trip) or sync fallback ---
#if USE_ASYNC_B
#pragma unroll
        for (int i = 0; i < 8; ++i) {
            __builtin_amdgcn_global_load_async_to_lds_b128(
                (__attribute__((address_space(1))) v4i_b128*)
                    (wbf + (size_t)bRow[i] * DIMK + k0 + bCol[i]),
                (__attribute__((address_space(3))) v4i_b128*)&Bbuf[bRow[i]][bCol[i]],
                0, 0);
        }
#else
        uint4 bTmp[8];
#pragma unroll
        for (int i = 0; i < 8; ++i)
            bTmp[i] = *(const uint4*)(wbf + (size_t)bRow[i] * DIMK + k0 + bCol[i]);
#endif

        // --- drain A staging registers into LDS (fp32 -> bf16) ---
#pragma unroll
        for (int i = 0; i < 8; ++i) {
            ushort4 r;
            r.x = f32_to_bf16_rne(aReg[i].x);
            r.y = f32_to_bf16_rne(aReg[i].y);
            r.z = f32_to_bf16_rne(aReg[i].z);
            r.w = f32_to_bf16_rne(aReg[i].w);
            *(ushort4*)&Abuf[aRow[i]][aCol[i]] = r;
        }
#if USE_ASYNC_B
        __builtin_amdgcn_s_wait_asynccnt(0);
#else
#pragma unroll
        for (int i = 0; i < 8; ++i)
            *(uint4*)&Bbuf[bRow[i]][bCol[i]] = bTmp[i];
#endif
        __syncthreads();

        // --- issue next A chunk's HBM loads; they fly under the WMMA block ---
        if (c + 1 < NCHUNK) {
            const int k0n = k0 + KC;
#pragma unroll
            for (int i = 0; i < 8; ++i)
                aReg[i] = *(const float4*)(x + (tokBase + aRow[i]) * DIMK + k0n + aCol[i]);
            // B for next chunk into WGP$ (near-scope prefetch, no VGPR cost)
            __builtin_prefetch(wbf + (size_t)bRow[0] * DIMK + k0n + bCol[0], 0, 3);
        }
        // x two chunks ahead into L2 (DEV-scope prefetch)
        if (c + 2 < NCHUNK) {
            __builtin_prefetch(x + (tokBase + aRow[0]) * DIMK + k0 + 2 * KC + aCol[0], 0, 1);
        }

        // --- compute: 2 k-steps of 32, 16 N-tiles each; B pipelined 1 deep ---
#pragma unroll
        for (int ks = 0; ks < 2; ++ks) {
            const int kb = ks * 32;
            Frag32B afr;
            afr.u[0] = *(const uint4*)&Abuf[wid * 16 + m16][kb + kh];
            afr.u[1] = *(const uint4*)&Abuf[wid * 16 + m16][kb + kh + 16];

            Frag32B bcur, bnext;
            bcur.u[0] = *(const uint4*)&Bbuf[m16][kb + kh];
            bcur.u[1] = *(const uint4*)&Bbuf[m16][kb + kh + 16];
#pragma unroll
            for (int t = 0; t < 16; ++t) {
                if (t + 1 < 16) {
                    const int e = (t + 1) * 16 + m16;
                    bnext.u[0] = *(const uint4*)&Bbuf[e][kb + kh];
                    bnext.u[1] = *(const uint4*)&Bbuf[e][kb + kh + 16];
                }
                acc[t] = __builtin_amdgcn_wmma_f32_16x16x32_bf16(
                    false, afr.v, false, bcur.v, (short)0, acc[t], false, false);
                bcur = bnext;   // SSA rotation; unrolling collapses the copies
            }
        }
        __syncthreads();
    }

    // --- store logits: C/D layout -> lanes 0-15 rows r, lanes 16-31 rows r+8 ---
#pragma unroll
    for (int t = 0; t < 16; ++t) {
#pragma unroll
        for (int r = 0; r < 8; ++r) {
            int mm = (lane < 16) ? r : (r + 8);
            int nn = t * 16 + m16;
            logits[(tokBase + wid * 16 + mm) * NEXP + nn] = acc[t][r];
        }
    }
}

// ---------------------------------------------------------------------------
// Kernel 3: per-token softmax + group-limited top-k. One wave32 per token;
// lane L owns experts [8L, 8L+8); group g = L/4 covers experts [32g, 32g+32).
// ---------------------------------------------------------------------------
__global__ void __launch_bounds__(256) router_topk_kernel(
        const float* __restrict__ logits,
        float* __restrict__ wout,
        int* __restrict__ iout) {
    const int lane = threadIdx.x & 31;
    const int wid  = threadIdx.x >> 5;
    const int t    = blockIdx.x * 8 + wid;

    const float4* lp = (const float4*)(logits + (size_t)t * NEXP + lane * 8);
    float4 a = lp[0], b = lp[1];
    float p[8] = {a.x, a.y, a.z, a.w, b.x, b.y, b.z, b.w};

    // softmax over 256 (wave reductions)
    float m = p[0];
#pragma unroll
    for (int j = 1; j < 8; ++j) m = fmaxf(m, p[j]);
#pragma unroll
    for (int off = 16; off >= 1; off >>= 1) m = fmaxf(m, __shfl_xor(m, off, 32));
    float s = 0.f;
#pragma unroll
    for (int j = 0; j < 8; ++j) { p[j] = __expf(p[j] - m); s += p[j]; }
#pragma unroll
    for (int off = 16; off >= 1; off >>= 1) s += __shfl_xor(s, off, 32);
    float inv = 1.0f / s;
#pragma unroll
    for (int j = 0; j < 8; ++j) p[j] *= inv;

    // group maxes: reduce within quads (group g = lanes 4g..4g+3)
    float gm = p[0];
#pragma unroll
    for (int j = 1; j < 8; ++j) gm = fmaxf(gm, p[j]);
    gm = fmaxf(gm, __shfl_xor(gm, 1, 32));
    gm = fmaxf(gm, __shfl_xor(gm, 2, 32));
    float garr[8];
#pragma unroll
    for (int g = 0; g < 8; ++g) garr[g] = __shfl(gm, g * 4, 32);

    // top-4 groups (redundant on every lane; lowest index wins ties like top_k)
    unsigned keep = 0;
#pragma unroll
    for (int it = 0; it < NLIM; ++it) {
        float best = -__builtin_inff();
        int bi = 0;
#pragma unroll
        for (int g = 0; g < NGROUPS; ++g)
            if (!((keep >> g) & 1u) && garr[g] > best) { best = garr[g]; bi = g; }
        keep |= 1u << bi;
    }

    // mask out non-kept groups, then 8 rounds of wave arg-max
    const bool kept = (keep >> (lane >> 2)) & 1u;
    float pl[8];
#pragma unroll
    for (int j = 0; j < 8; ++j) pl[j] = kept ? p[j] : -__builtin_inff();

    float myv = 0.f;
    int   myi = 0;
    float wsum = 0.f;
#pragma unroll
    for (int r = 0; r < TOPK; ++r) {
        float lv = -__builtin_inff();
        int lj = 0;
#pragma unroll
        for (int j = 0; j < 8; ++j)
            if (pl[j] > lv) { lv = pl[j]; lj = j; }
        int li = lane * 8 + lj;
#pragma unroll
        for (int off = 1; off < 32; off <<= 1) {
            float ov = __shfl_xor(lv, off, 32);
            int   oi = __shfl_xor(li, off, 32);
            if (ov > lv || (ov == lv && oi < li)) { lv = ov; li = oi; }
        }
        wsum += lv;
        if (lane == r) { myv = lv; myi = li; }
        if ((li >> 3) == lane) pl[li & 7] = -__builtin_inff();
    }

    if (lane < TOPK) {
        float wnorm = myv / (wsum + 1e-9f);   // ROUTE_SCALE == 1.0
        wout[(size_t)t * TOPK + lane] = wnorm;
        iout[(size_t)t * TOPK + lane] = myi;
    }
}

// ---------------------------------------------------------------------------
extern "C" void kernel_launch(void* const* d_in, const int* in_sizes, int n_in,
                              void* d_out, int out_size, void* d_ws, size_t ws_size,
                              hipStream_t stream) {
    const float* x = (const float*)d_in[0];                 // [16384, 4096] f32
    const float* w = (const float*)d_in[1];                 // [256, 4096] f32

    unsigned short* wbf = (unsigned short*)d_ws;            // 2 MB bf16 weight
    float* logits = (float*)((char*)d_ws +
                             (size_t)NEXP * DIMK * sizeof(unsigned short)); // 16 MB

    float* wout = (float*)d_out;                            // [16384, 8] f32
    int*   iout = (int*)d_out + (size_t)TOKENS * TOPK;      // [16384, 8] i32

    cvt_weight_kernel<<<(NEXP * DIMK) / (256 * 4), 256, 0, stream>>>(w, wbf);
    router_gemm_kernel<<<TOKENS / 128, 256, 0, stream>>>(x, wbf, logits);
    router_topk_kernel<<<TOKENS / 8, 256, 0, stream>>>(logits, wout, iout);
}
